// Head_77704548319884
// MI455X (gfx1250) — matrix operand order repaired and checked by reading, not compiled, not run on hardware
//
#include <hip/hip_runtime.h>
#include <hip/hip_bf16.h>
#include <stdint.h>

// Causal single-head attention, B=16 T=2048 C=1024 H=128, fp32 in/out.
// bf16 WMMA (V_WMMA_F32_16X16X32_BF16) with f32 accumulation.
//  Kernel 0: transpose/convert W -> Wt[3][H][C] bf16 (B-fragment friendly).
//  Kernel 1: Q/K/V projections -> bf16 workspace; V stored transposed [B][H][T].
//  Kernel 2: flash-attention per (batch, 64-query tile), 4 waves x 16 rows.
// All WMMA fragments are per-lane contiguous 16B chunks -> ds/global _b128.
// K/V tile staging uses GLOBAL_LOAD_ASYNC_TO_LDS_B128 (ASYNCcnt) when available.

typedef __bf16 bf16_t;
typedef __attribute__((ext_vector_type(16))) __bf16 v16bf;
typedef __attribute__((ext_vector_type(8)))  float  v8f;
typedef __attribute__((ext_vector_type(4)))  int    v4i;

#define B_SZ 16
#define T_SZ 2048
#define C_SZ 1024
#define H_SZ 128

__device__ __forceinline__ uint16_t bfbits(float f) {
  union { float f; uint32_t u; } c; c.f = f;
  return (uint16_t)((c.u + 0x7FFFu + ((c.u >> 16) & 1u)) >> 16);  // RNE
}
__device__ __forceinline__ bf16_t f2bf(float f) {
  union { uint16_t s; bf16_t b; } o; o.s = bfbits(f);
  return o.b;
}
__device__ __forceinline__ uint32_t pk2bf(float lo, float hi) {
  return (uint32_t)bfbits(lo) | ((uint32_t)bfbits(hi) << 16);
}

union U16bf { uint4 q[2]; v16bf v; };

// A fragment (16x32 bf16): per-lane 16 contiguous K (K = e + 16*half)
__device__ __forceinline__ v16bf load_a2x16(const bf16_t* p) {
  U16bf u;
  u.q[0] = *(const uint4*)(p);
  u.q[1] = *(const uint4*)(p + 8);
  return u.v;
}
// B fragment (32x16 bf16): per-lane 16 contiguous K (K = e + 16*half)
__device__ __forceinline__ v16bf load_b32B(const bf16_t* p) {
  U16bf u;
  u.q[0] = *(const uint4*)(p);
  u.q[1] = *(const uint4*)(p + 8);
  return u.v;
}

// ---- async global->LDS staging (CDNA5 GLOBAL_LOAD_ASYNC_TO_LDS_B128) -------
#if __has_builtin(__builtin_amdgcn_global_load_async_to_lds_b128)
#define ASYNC_LDS 1
typedef __attribute__((address_space(1))) v4i* gv4i_p;   // global int4*
typedef __attribute__((address_space(3))) v4i* lv4i_p;   // LDS int4*
__device__ __forceinline__ void async_cp16(const void* g, void* l) {
  __builtin_amdgcn_global_load_async_to_lds_b128((gv4i_p)g, (lv4i_p)l, 0, 0);
}
__device__ __forceinline__ void wait_async0() {
#if __has_builtin(__builtin_amdgcn_s_wait_asynccnt)
  __builtin_amdgcn_s_wait_asynccnt(0);
#else
  asm volatile("s_wait_asynccnt 0" ::: "memory");
#endif
}
#endif

// ---------------------------------------------------------------------------
// Kernel 0: Wt[sel][H][C] bf16 = transpose(W[C][H] fp32). 393216 elements.
// ---------------------------------------------------------------------------
__global__ __launch_bounds__(256)
void wtrans_kernel(const float* __restrict__ Wk, const float* __restrict__ Wq,
                   const float* __restrict__ Wv, bf16_t* __restrict__ Wt) {
  int idx = blockIdx.x * 256 + threadIdx.x;          // over 3*H*C
  int sel = idx / (H_SZ * C_SZ);
  int rem = idx % (H_SZ * C_SZ);
  int c = rem / H_SZ, h = rem % H_SZ;                // coalesced reads over h
  const float* W = (sel == 0) ? Wk : (sel == 1) ? Wq : Wv;
  Wt[((size_t)sel * H_SZ + h) * C_SZ + c] = f2bf(W[(size_t)c * H_SZ + h]);
}

// ---------------------------------------------------------------------------
// Kernel 1: projections. grid = (BT/128, 3), block = 256 (8 waves).
// x tile double-buffered in LDS; W B-fragments straight from global Wt.
// ---------------------------------------------------------------------------
__global__ __launch_bounds__(256)
void proj_gemm_kernel(const float* __restrict__ x,
                      const bf16_t* __restrict__ Wt,
                      bf16_t* __restrict__ Kb,
                      bf16_t* __restrict__ Qb,
                      bf16_t* __restrict__ Vt) {
  __shared__ bf16_t xs[2][128][40];   // 128 rows x 32 K, 80B rows (16B aligned)

  const int tid  = threadIdx.x;
  const int wave = tid >> 5;
  const int lane = tid & 31;
  const int half = lane >> 4;
  const int l16  = lane & 15;

  const int m0  = blockIdx.x * 128;
  const int sel = blockIdx.y;
  const bf16_t* Wsel = Wt + (size_t)sel * H_SZ * C_SZ;

  v8f acc[8] = {};

  for (int kc = 0; kc < C_SZ / 32; ++kc) {
    const int buf = kc & 1;
    // stage x tile (128x32) fp32 -> bf16, vectorized: 1024 float4 / 256 thr
    #pragma unroll
    for (int i = 0; i < 4; ++i) {
      int idx = i * 256 + tid;
      int r = idx >> 3, cq = idx & 7;
      float4 f = ((const float4*)(x + (size_t)(m0 + r) * C_SZ + kc * 32))[cq];
      uint2 pk;
      pk.x = pk2bf(f.x, f.y);
      pk.y = pk2bf(f.z, f.w);
      *(uint2*)&xs[buf][r][cq * 4] = pk;
    }
    if (kc + 1 < C_SZ / 32 && tid < 128)   // prefetch next x chunk (1 line/row)
      __builtin_prefetch(x + (size_t)(m0 + tid) * C_SZ + (kc + 1) * 32, 0, 1);
    __syncthreads();

    v16bf a = load_a2x16(&xs[buf][wave * 16 + l16][16 * half]);
    // preload all 8 B fragments -> one load clause, then 8 independent WMMAs
    v16bf bfr[8];
    #pragma unroll
    for (int t = 0; t < 8; ++t)
      bfr[t] = load_b32B(Wsel + (size_t)(t * 16 + l16) * C_SZ + kc * 32 + 16 * half);
    #pragma unroll
    for (int t = 0; t < 8; ++t)
      acc[t] = __builtin_amdgcn_wmma_f32_16x16x32_bf16(
          false, a, false, bfr[t], (short)0, acc[t], false, false);
  }

  // store (C/D layout: row = j + 8*half, col = t*16 + lane&15)
  if (sel == 2) {   // V stored transposed: Vt[b][h][t]
    const int bb = m0 / T_SZ;
    #pragma unroll
    for (int t = 0; t < 8; ++t)
      #pragma unroll
      for (int j = 0; j < 8; ++j) {
        int row  = m0 + wave * 16 + j + 8 * half;
        int col  = t * 16 + l16;
        Vt[((size_t)bb * H_SZ + col) * T_SZ + (row % T_SZ)] = f2bf(acc[t][j]);
      }
  } else {
    bf16_t* Ob = (sel == 0) ? Kb : Qb;
    #pragma unroll
    for (int t = 0; t < 8; ++t)
      #pragma unroll
      for (int j = 0; j < 8; ++j) {
        int row = m0 + wave * 16 + j + 8 * half;
        int col = t * 16 + l16;
        Ob[(size_t)row * H_SZ + col] = f2bf(acc[t][j]);
      }
  }
}

// ---------------------------------------------------------------------------
// Kernel 2: flash attention. grid = (T/64, B), block = 128 (4 waves).
// ---------------------------------------------------------------------------
__global__ __launch_bounds__(128)
void attn_kernel(const bf16_t* __restrict__ Qb,
                 const bf16_t* __restrict__ Kb,
                 const bf16_t* __restrict__ Vt,
                 float* __restrict__ out) {
  __shared__ bf16_t Ks[64][136];     // [key][h]   rows 272B (17*16)
  __shared__ bf16_t VsT[128][72];    // [h][key]   rows 144B (9*16)
  __shared__ float  Sb[4][16][68];   // per-wave 16x64 scores, rows 272B
  __shared__ bf16_t Pb[4][16][72];   // per-wave 16x64 probs,  rows 144B
  __shared__ float  rowm[4][16], rowl[4][16], ralpha[4][16];

  const int tid  = threadIdx.x;
  const int wave = tid >> 5;
  const int lane = tid & 31;
  const int half = lane >> 4;
  const int l16  = lane & 15;

  const int q0 = blockIdx.x * 64;
  const int b  = blockIdx.y;
  const size_t bt0 = (size_t)b * T_SZ;
  const float sscale = 0.08838834764831845f;   // H^-0.5

  // Q fragments: 4 x A[16x32], 2 x global_load_b128 each
  v16bf qf[4];
  {
    const bf16_t* qrow = Qb + (bt0 + q0 + wave * 16 + l16) * H_SZ;
    #pragma unroll
    for (int f = 0; f < 4; ++f)
      qf[f] = load_a2x16(qrow + f * 32 + 16 * half);
  }

  v8f o[8] = {};
  if (lane < 16) { rowm[wave][lane] = -1e30f; rowl[wave][lane] = 0.f; }

  const int ntiles = q0 / 64 + 1;              // causal lower triangle only
  for (int kt = 0; kt < ntiles; ++kt) {
    const int kv0 = kt * 64;
    if (kt + 1 < ntiles) {                     // prefetch next K/V tiles
      __builtin_prefetch(Kb + (bt0 + kv0 + 64) * H_SZ + tid * 64, 0, 1);
      __builtin_prefetch(Vt + ((size_t)b * H_SZ + tid) * T_SZ + kv0 + 64, 0, 1);
    }
    __syncthreads();                           // protect Ks/VsT reuse

    // stage K tile [64 keys][128 h] and V^T tile [128 h][64 keys]
    {
      const uint4* gk = (const uint4*)(Kb + (bt0 + kv0) * H_SZ);
#if defined(ASYNC_LDS)
      #pragma unroll
      for (int i = 0; i < 8; ++i) {
        int idx = i * 128 + tid;               // 1024 x 16B
        int r = idx >> 4, cv = idx & 15;
        async_cp16(gk + r * 16 + cv, &Ks[r][cv * 8]);
      }
      #pragma unroll
      for (int i = 0; i < 8; ++i) {
        int idx = i * 128 + tid;
        int r = idx >> 3, cv = idx & 7;        // r = h row, 8 x 16B per row
        async_cp16(Vt + ((size_t)b * H_SZ + r) * T_SZ + kv0 + cv * 8,
                   &VsT[r][cv * 8]);
      }
      wait_async0();
#else
      #pragma unroll
      for (int i = 0; i < 8; ++i) {
        int idx = i * 128 + tid;
        int r = idx >> 4, cv = idx & 15;
        *(uint4*)&Ks[r][cv * 8] = gk[r * 16 + cv];
      }
      #pragma unroll
      for (int i = 0; i < 8; ++i) {
        int idx = i * 128 + tid;
        int r = idx >> 3, cv = idx & 7;
        *(uint4*)&VsT[r][cv * 8] =
            *(const uint4*)(Vt + ((size_t)b * H_SZ + r) * T_SZ + kv0 + cv * 8);
      }
#endif
    }
    __syncthreads();

    // S = Q @ K^T : per k-step preload 4 B fragments, then 4 WMMAs
    v8f s[4] = {};
    #pragma unroll
    for (int kc = 0; kc < 4; ++kc) {
      v16bf bfr[4];
      #pragma unroll
      for (int nt = 0; nt < 4; ++nt)
        bfr[nt] = load_b32B(&Ks[nt * 16 + l16][kc * 32 + 16 * half]);
      #pragma unroll
      for (int nt = 0; nt < 4; ++nt)
        s[nt] = __builtin_amdgcn_wmma_f32_16x16x32_bf16(
            false, qf[kc], false, bfr[nt], (short)0, s[nt], false, false);
    }

    // spill scaled S
    #pragma unroll
    for (int nt = 0; nt < 4; ++nt)
      #pragma unroll
      for (int j = 0; j < 8; ++j)
        Sb[wave][j + 8 * half][nt * 16 + l16] = s[nt][j] * sscale;
    asm volatile("s_wait_dscnt 0" ::: "memory");

    // online softmax: lane -> row l16, columns [half*32, +32)
    {
      const int r  = l16;
      const int qg = q0 + wave * 16 + r;
      const int c0 = half * 32;
      const bool diag = (kt == ntiles - 1);
      float4 vv[8];
      float mx = -1e30f;
      #pragma unroll
      for (int i = 0; i < 8; ++i) {
        float4 sv = *(const float4*)&Sb[wave][r][c0 + 4 * i];
        if (diag) {
          int kg = kv0 + c0 + 4 * i;
          if (kg + 0 > qg) sv.x = -1e30f;
          if (kg + 1 > qg) sv.y = -1e30f;
          if (kg + 2 > qg) sv.z = -1e30f;
          if (kg + 3 > qg) sv.w = -1e30f;
        }
        vv[i] = sv;
        mx = fmaxf(mx, fmaxf(fmaxf(sv.x, sv.y), fmaxf(sv.z, sv.w)));
      }
      mx = fmaxf(mx, __shfl_xor(mx, 16, 32));
      float mold = rowm[wave][r];
      float mnew = fmaxf(mold, mx);
      float al   = __expf(mold - mnew);
      float sum  = 0.f;
      #pragma unroll
      for (int i = 0; i < 8; ++i) {
        float p0 = __expf(vv[i].x - mnew), p1 = __expf(vv[i].y - mnew);
        float p2 = __expf(vv[i].z - mnew), p3 = __expf(vv[i].w - mnew);
        sum += (p0 + p1) + (p2 + p3);
        uint2 pk; pk.x = pk2bf(p0, p1); pk.y = pk2bf(p2, p3);
        *(uint2*)&Pb[wave][r][c0 + 4 * i] = pk;
      }
      sum += __shfl_xor(sum, 16, 32);
      if (half == 0) {
        rowm[wave][r]   = mnew;
        rowl[wave][r]   = rowl[wave][r] * al + sum;
        ralpha[wave][r] = al;
      }
    }
    asm volatile("s_wait_dscnt 0" ::: "memory");

    // rescale running O
    #pragma unroll
    for (int j = 0; j < 8; ++j) {
      float al = ralpha[wave][j + 8 * half];
      #pragma unroll
      for (int t = 0; t < 8; ++t) o[t][j] *= al;
    }

    // O += P @ V : A from Pb (2x b128), B fragments preloaded in groups of 4
    #pragma unroll
    for (int kc = 0; kc < 2; ++kc) {
      v16bf pf = load_a2x16(&Pb[wave][l16][kc * 32 + 16 * half]);
      #pragma unroll
      for (int g = 0; g < 2; ++g) {
        v16bf vf[4];
        #pragma unroll
        for (int n4 = 0; n4 < 4; ++n4)
          vf[n4] = load_b32B(&VsT[(g * 4 + n4) * 16 + l16][kc * 32 + 16 * half]);
        #pragma unroll
        for (int n4 = 0; n4 < 4; ++n4)
          o[g * 4 + n4] = __builtin_amdgcn_wmma_f32_16x16x32_bf16(
              false, pf, false, vf[n4], (short)0, o[g * 4 + n4], false, false);
      }
    }
  }

  asm volatile("s_wait_dscnt 0" ::: "memory");

  // final 1/l normalization, fp32 store
  #pragma unroll
  for (int j = 0; j < 8; ++j) {
    float inv = 1.f / rowl[wave][j + 8 * half];
    int row = q0 + wave * 16 + j + 8 * half;
    float* orow = out + (bt0 + row) * H_SZ;
    #pragma unroll
    for (int t = 0; t < 8; ++t)
      orow[t * 16 + l16] = o[t][j] * inv;
  }
}

// ---------------------------------------------------------------------------
extern "C" void kernel_launch(void* const* d_in, const int* in_sizes, int n_in,
                              void* d_out, int out_size, void* d_ws, size_t ws_size,
                              hipStream_t stream) {
  (void)in_sizes; (void)n_in; (void)out_size; (void)ws_size;
  const float* x  = (const float*)d_in[0];
  const float* Wk = (const float*)d_in[1];
  const float* Wq = (const float*)d_in[2];
  const float* Wv = (const float*)d_in[3];
  float* out = (float*)d_out;

  const size_t BT = (size_t)B_SZ * T_SZ;
  // workspace (bf16): K [BT][H], Q [BT][H], V^T [B][H][T], W^T [3][H][C]
  bf16_t* Kb = (bf16_t*)d_ws;
  bf16_t* Qb = Kb + BT * H_SZ;
  bf16_t* Vt = Qb + BT * H_SZ;
  bf16_t* Wt = Vt + BT * H_SZ;

  wtrans_kernel<<<dim3(3 * H_SZ * C_SZ / 256), dim3(256), 0, stream>>>(Wk, Wq, Wv, Wt);

  proj_gemm_kernel<<<dim3((unsigned)(BT / 128), 3), dim3(256), 0, stream>>>(
      x, Wt, Kb, Qb, Vt);

  attn_kernel<<<dim3(T_SZ / 64, B_SZ), dim3(128), 0, stream>>>(Qb, Kb, Vt, out);
}
